// GraphLearning_27281632264514
// MI455X (gfx1250) — matrix operand order repaired and checked by reading, not compiled, not run on hardware
//
#include <hip/hip_runtime.h>
#include <hip/hip_bf16.h>

#define NODES 50000
#define EDGES 800000
#define FIN   256
#define CD    32
#define HEADS 4
#define HCD   128
#define LAYERS 2
#define EPSV  1e-5f
#define NEG_SLOPE 0.2f

typedef __attribute__((ext_vector_type(16))) __bf16 v16bf;
typedef __attribute__((ext_vector_type(8)))  __bf16 v8bf;
typedef __attribute__((ext_vector_type(8)))  float  v8f;

// ---------------- utility device functions ----------------

__device__ __forceinline__ float wave_sum_f(float v) {
#pragma unroll
  for (int off = 16; off > 0; off >>= 1) v += __shfl_xor(v, off, 32);
  return v;
}

__device__ __forceinline__ double wave_sum_d(double v) {
#pragma unroll
  for (int off = 16; off > 0; off >>= 1) v += __shfl_xor(v, off, 32);
  return v;
}

// order-preserving bijection float -> u32 (for atomic max on floats)
__device__ __forceinline__ unsigned ord_enc(float f) {
  unsigned u = __float_as_uint(f);
  return (u & 0x80000000u) ? ~u : (u | 0x80000000u);
}
__device__ __forceinline__ float ord_dec(unsigned e) {
  unsigned u = (e & 0x80000000u) ? (e ^ 0x80000000u) : ~e;
  return __uint_as_float(u);
}

// ---------------- elementwise helpers ----------------

__global__ void f32_to_bf16_kernel(const float* __restrict__ in, __bf16* __restrict__ out, long n) {
  long i = (long)blockIdx.x * blockDim.x + threadIdx.x;
  if (i < n) out[i] = (__bf16)in[i];
}

__global__ void fill_f32_kernel(float* __restrict__ p, float v, long n) {
  long i = (long)blockIdx.x * blockDim.x + threadIdx.x;
  if (i < n) p[i] = v;
}

__global__ void fill_u32_kernel(unsigned* __restrict__ p, unsigned v, long n) {
  long i = (long)blockIdx.x * blockDim.x + threadIdx.x;
  if (i < n) p[i] = v;
}

// ---------------- B-operand pre-pack: f32 row-major [K,Nc] -> bf16 WMMA fragment layout -------
// Packed layout: dst[((kb2*tilesN + tn)*32 + lane)*16 + j] where
//   k = kb2*32 + (lane&16 ? 16 : 0) + j ; n = tn*16 + (lane&15)
// so each lane's 16 B-fragment values are contiguous (32 B) and lanes are consecutive.
__global__ void pack_b_kernel(const float* __restrict__ src, __bf16* __restrict__ dst,
                              int K, int Nc) {
  int idx = blockIdx.x * blockDim.x + threadIdx.x;
  int tilesN = Nc >> 4;
  int total = (K >> 5) * tilesN * 512;
  if (idx >= total) return;
  int j    = idx & 15;
  int lane = (idx >> 4) & 31;
  int rest = idx >> 9;            // kb2*tilesN + tn
  int tn   = rest % tilesN;
  int kb2  = rest / tilesN;
  int k = (kb2 << 5) + ((lane & 16) ? 16 : 0) + j;
  int n = (tn << 4) + (lane & 15);
  dst[idx] = (__bf16)src[(long)k * Nc + n];
}

// ---------------- WMMA GEMM: C[M,Nc] = A[M,K](bf16) * Bpacked(bf16) + bias ----------------
// One wave computes MT=4 vertically-stacked 16x16 output tiles sharing each B fragment.
// A 16x32 bf16 fragment (this code's contiguous k order, same as pack_b): lane L holds row L%16;
//   elems 0..7 -> K = ka..ka+7, elems 8..15 -> K = ka+8..ka+15, ka = k0 + (L<16?0:16).
// Both runs are contiguous & 16B-aligned -> two global_load_b128 per fragment.
__global__ void wmma_gemm_bias_kernel(const __bf16* __restrict__ A, const __bf16* __restrict__ Bp,
                                      const float* __restrict__ bias, float* __restrict__ C,
                                      int M, int K, int Nc) {
  const int MT = 4;
  int wave = blockIdx.x * (blockDim.x >> 5) + (threadIdx.x >> 5);
  int tilesN = Nc >> 4;
  int tilesM = M >> 4;
  int mGroups = (tilesM + MT - 1) / MT;
  if (wave >= mGroups * tilesN) return;         // wave-uniform -> EXEC all-1 inside WMMA
  int tn = wave % tilesN;
  int tmBase = (wave / tilesN) * MT;

  int lane = threadIdx.x & 31;
  int half = (lane >> 4) & 1;
  int rlo  = lane & 15;
  int col  = (tn << 4) + rlo;

  v8f acc[MT];
#pragma unroll
  for (int mt = 0; mt < MT; ++mt) acc[mt] = (v8f){};

  for (int k0 = 0; k0 < K; k0 += 32) {
    // B fragment: one contiguous 32B read per lane from the packed layout
    v16bf b = *(const v16bf*)(Bp + ((((size_t)(k0 >> 5) * tilesN + tn) << 9) + (lane << 4)));
    int ka = k0 + half * 16;
#pragma unroll
    for (int mt = 0; mt < MT; ++mt) {
      int tm = tmBase + mt;
      if (tm >= tilesM) break;                  // uniform tail guard
      const __bf16* __restrict__ Arow = A + (size_t)(tm * 16 + rlo) * K;
      union { v16bf v; struct { v8bf lo; v8bf hi; } s; } au;
      au.s.lo = *(const v8bf*)(Arow + ka);
      au.s.hi = *(const v8bf*)(Arow + ka + 8);
      acc[mt] = __builtin_amdgcn_wmma_f32_16x16x32_bf16(false, au.v, false, b, (short)0,
                                                        acc[mt], false, false);
    }
  }

  float bv = bias ? bias[col] : 0.0f;
#pragma unroll
  for (int mt = 0; mt < MT; ++mt) {
    int tm = tmBase + mt;
    if (tm >= tilesM) break;
    int rbase = tm * 16 + half * 8;
#pragma unroll
    for (int r = 0; r < 8; ++r)
      C[(size_t)(rbase + r) * Nc + col] = acc[mt][r] + bv;
  }
}

// ---------------- node LayerNorm (last dim = 32) + ReLU + JK-max, one wave per node ------------

__global__ void node_ln_relu_max_kernel(const float* __restrict__ pre,
                                        const float* __restrict__ w, const float* __restrict__ b,
                                        float* __restrict__ h, __bf16* __restrict__ hbf,
                                        float* __restrict__ jk, int n, int init) {
  int node = blockIdx.x * (blockDim.x >> 5) + (threadIdx.x >> 5);
  if (node >= n) return;
  int c = threadIdx.x & 31;
  long o = (long)node * CD + c;
  float v = pre[o];
  float mean = wave_sum_f(v) * (1.0f / CD);
  float xc = v - mean;
  float var = wave_sum_f(xc * xc) * (1.0f / CD);
  float y = xc * rsqrtf(var + EPSV) * w[c] + b[c];
  y = fmaxf(y, 0.0f);
  h[o] = y;
  hbf[o] = (__bf16)y;
  jk[o] = init ? y : fmaxf(jk[o], y);
}

// ---------------- edge pass 1: logits + segment max (thread per edge,head) ----------------

__global__ void edge_logits_kernel(const float* __restrict__ xl, const float* __restrict__ xr,
                                   const float* __restrict__ ea,
                                   const int* __restrict__ srcIdx, const int* __restrict__ dstIdx,
                                   const float* __restrict__ WeL, const float* __restrict__ attL,
                                   float* __restrict__ logits, unsigned* __restrict__ mbuf) {
  int t = blockIdx.x * blockDim.x + threadIdx.x;
  int tot = (EDGES + NODES) * HEADS;
  if (t >= tot) return;
  int e = t >> 2, hh = t & 3;
  int s, d; float wv;
  if (e < EDGES) { s = srcIdx[e]; d = dstIdx[e]; wv = ea[e]; }
  else           { s = e - EDGES; d = s;          wv = 0.0f; }

  const float4* __restrict__ pl = (const float4*)(xl + (long)s * HCD + hh * CD);
  const float4* __restrict__ pr = (const float4*)(xr + (long)d * HCD + hh * CD);
  __builtin_prefetch(pl, 0, 3);
  __builtin_prefetch(pr, 0, 3);
  const float4* __restrict__ pw = (const float4*)(WeL + hh * CD);
  const float4* __restrict__ pa = (const float4*)(attL + hh * CD);

  float acc = 0.0f;
#pragma unroll
  for (int q = 0; q < 8; ++q) {
    float4 a4 = pl[q], b4 = pr[q], w4 = pw[q], t4 = pa[q];
    float zv;
    zv = a4.x + b4.x + wv * w4.x; zv = zv > 0.f ? zv : NEG_SLOPE * zv; acc += zv * t4.x;
    zv = a4.y + b4.y + wv * w4.y; zv = zv > 0.f ? zv : NEG_SLOPE * zv; acc += zv * t4.y;
    zv = a4.z + b4.z + wv * w4.z; zv = zv > 0.f ? zv : NEG_SLOPE * zv; acc += zv * t4.z;
    zv = a4.w + b4.w + wv * w4.w; zv = zv > 0.f ? zv : NEG_SLOPE * zv; acc += zv * t4.w;
  }
  logits[(long)e * HEADS + hh] = acc;
  atomicMax(mbuf + (long)d * HEADS + hh, ord_enc(acc));
}

// ---------------- edge pass 2: exp(logit - max) + segment sum ----------------

__global__ void edge_exp_kernel(float* __restrict__ logits, const unsigned* __restrict__ mbuf,
                                const int* __restrict__ dstIdx, float* __restrict__ sbuf) {
  int t = blockIdx.x * blockDim.x + threadIdx.x;
  int tot = (EDGES + NODES) * HEADS;
  if (t >= tot) return;
  int e = t >> 2, hh = t & 3;
  int d = (e < EDGES) ? dstIdx[e] : (e - EDGES);
  float m = ord_dec(mbuf[(long)d * HEADS + hh]);
  float ex = __expf(logits[(long)e * HEADS + hh] - m);
  logits[(long)e * HEADS + hh] = ex;
  atomicAdd(sbuf + (long)d * HEADS + hh, ex);
}

// ---------------- edge pass 3: alpha * xl[src] scatter-add ----------------

__global__ void edge_scatter_kernel(const float* __restrict__ exv, const float* __restrict__ sbuf,
                                    const float* __restrict__ xl,
                                    const int* __restrict__ srcIdx, const int* __restrict__ dstIdx,
                                    float* __restrict__ agg) {
  int t = blockIdx.x * blockDim.x + threadIdx.x;
  int tot = (EDGES + NODES) * HEADS;
  if (t >= tot) return;
  int e = t >> 2, hh = t & 3;
  int s, d;
  if (e < EDGES) { s = srcIdx[e]; d = dstIdx[e]; }
  else           { s = e - EDGES; d = s; }
  const float4* __restrict__ pl = (const float4*)(xl + (long)s * HCD + hh * CD);
  __builtin_prefetch(pl, 0, 3);
  float alpha = exv[(long)e * HEADS + hh] / sbuf[(long)d * HEADS + hh];
  float* __restrict__ po = agg + (long)d * HCD + hh * CD;
#pragma unroll
  for (int q = 0; q < 8; ++q) {
    float4 v = pl[q];
    atomicAdd(po + q * 4 + 0, alpha * v.x);
    atomicAdd(po + q * 4 + 1, alpha * v.y);
    atomicAdd(po + q * 4 + 2, alpha * v.z);
    atomicAdd(po + q * 4 + 3, alpha * v.w);
  }
}

// ---------------- graph LayerNorm: global sum/sumsq reduction then normalize+ReLU ----------------

__global__ void reduce_sumsq_kernel(const float* __restrict__ agg, const float* __restrict__ gatb,
                                    double* __restrict__ red, long total) {
  long i = (long)blockIdx.x * blockDim.x + threadIdx.x;
  long stride = (long)gridDim.x * blockDim.x;
  double ls = 0.0, lq = 0.0;
  for (; i < total; i += stride) {
    float v = agg[i] + gatb[i & (HCD - 1)];
    ls += v;
    lq += (double)v * (double)v;
  }
  ls = wave_sum_d(ls);
  lq = wave_sum_d(lq);
  if ((threadIdx.x & 31) == 0) {
    atomicAdd(&red[0], ls);
    atomicAdd(&red[1], lq);
  }
}

__global__ void graph_ln_relu_kernel(const float* __restrict__ agg, const float* __restrict__ gatb,
                                     const double* __restrict__ red,
                                     const float* __restrict__ w, const float* __restrict__ b,
                                     __bf16* __restrict__ gbf, long total) {
  long i = (long)blockIdx.x * blockDim.x + threadIdx.x;
  if (i >= total) return;
  double mu = red[0] / (double)total;
  double var = red[1] / (double)total - mu * mu;
  float inv = rsqrtf((float)var + EPSV);
  int c = (int)(i & (HCD - 1));
  float v = agg[i] + gatb[c];
  float g = (v - (float)mu) * inv * w[c] + b[c];
  gbf[i] = (__bf16)fmaxf(g, 0.0f);
}

// ---------------- host-side orchestration ----------------

extern "C" void kernel_launch(void* const* d_in, const int* in_sizes, int n_in,
                              void* d_out, int out_size, void* d_ws, size_t ws_size,
                              hipStream_t stream) {
  const float* x     = (const float*)d_in[0];
  const int*   eidx  = (const int*)  d_in[1];   // [2,E]: src then dst
  const float* eattr = (const float*)d_in[2];
  const float* W_emb = (const float*)d_in[3];
  const float* b_emb = (const float*)d_in[4];
  const float* ln0_w = (const float*)d_in[5];
  const float* ln0_b = (const float*)d_in[6];
  const float* Wl    = (const float*)d_in[7];
  const float* bl    = (const float*)d_in[8];
  const float* Wr    = (const float*)d_in[9];
  const float* br    = (const float*)d_in[10];
  const float* We    = (const float*)d_in[11];
  const float* att   = (const float*)d_in[12];
  const float* gat_b = (const float*)d_in[13];
  const float* ln1_w = (const float*)d_in[14];
  const float* ln1_b = (const float*)d_in[15];
  const float* lin_w = (const float*)d_in[16];
  const float* lin_b = (const float*)d_in[17];
  const float* ln2_w = (const float*)d_in[18];
  const float* ln2_b = (const float*)d_in[19];
  float* out = (float*)d_out;   // [N, C]; doubles as the JK-max accumulator

  char* ws = (char*)d_ws;
  size_t off = 0;
  auto alloc = [&](size_t bytes) -> void* {
    void* p = ws + off;
    off += (bytes + 255) & ~(size_t)255;
    return p;
  };

  __bf16* x_bf    = (__bf16*)alloc((size_t)NODES * FIN * 2);
  __bf16* h_bf    = (__bf16*)alloc((size_t)NODES * CD * 2);
  __bf16* g_bf    = (__bf16*)alloc((size_t)NODES * HCD * 2);
  __bf16* wemb_pk = (__bf16*)alloc((size_t)FIN * CD * 2);   // packed B operands
  __bf16* wl_pk   = (__bf16*)alloc((size_t)CD * HCD * 2);
  __bf16* wr_pk   = (__bf16*)alloc((size_t)CD * HCD * 2);
  __bf16* lw_pk   = (__bf16*)alloc((size_t)HCD * CD * 2);
  float* tmpC     = (float*)alloc((size_t)NODES * CD * 4);
  float* h        = (float*)alloc((size_t)NODES * CD * 4);
  float* xl       = (float*)alloc((size_t)NODES * HCD * 4);
  float* xr       = (float*)alloc((size_t)NODES * HCD * 4);
  float* agg      = (float*)alloc((size_t)NODES * HCD * 4);
  float* logits   = (float*)alloc((size_t)(EDGES + NODES) * HEADS * 4);
  unsigned* mbuf  = (unsigned*)alloc((size_t)NODES * HEADS * 4);
  float* sbuf     = (float*)alloc((size_t)NODES * HEADS * 4);
  double* red     = (double*)alloc(2 * sizeof(double));

  const int* srcIdx = eidx;
  const int* dstIdx = eidx + EDGES;

  const int T = 256;
  auto blocksFor = [](long n, int t) { return (int)((n + t - 1) / t); };
  auto gemmBlocks = [](int M, int Nc) {
    int waves = ((M / 16 + 3) / 4) * (Nc / 16);
    return (waves + 7) / 8;
  };
  const int edgeThreads = (EDGES + NODES) * HEADS;
  const unsigned ENC_NEG_INF = 0x007FFFFFu;   // ord_enc(-inf) = ~bits(-inf)

  // ---- embed: x(bf16) @ W_emb(packed bf16) + b ; node LN ; ReLU ; JK init ----
  f32_to_bf16_kernel<<<blocksFor((long)NODES * FIN, T), T, 0, stream>>>(x, x_bf, (long)NODES * FIN);
  pack_b_kernel<<<blocksFor((long)FIN * CD, T), T, 0, stream>>>(W_emb, wemb_pk, FIN, CD);
  wmma_gemm_bias_kernel<<<gemmBlocks(NODES, CD), T, 0, stream>>>(x_bf, wemb_pk, b_emb, tmpC,
                                                                 NODES, FIN, CD);
  node_ln_relu_max_kernel<<<(NODES + 7) / 8, T, 0, stream>>>(tmpC, ln0_w, ln0_b, h, h_bf, out,
                                                             NODES, 1);

  for (int i = 0; i < LAYERS; ++i) {
    pack_b_kernel<<<blocksFor((long)CD * HCD, T), T, 0, stream>>>(Wl + (size_t)i * CD * HCD,
                                                                  wl_pk, CD, HCD);
    pack_b_kernel<<<blocksFor((long)CD * HCD, T), T, 0, stream>>>(Wr + (size_t)i * CD * HCD,
                                                                  wr_pk, CD, HCD);
    pack_b_kernel<<<blocksFor((long)HCD * CD, T), T, 0, stream>>>(lin_w + (size_t)i * HCD * CD,
                                                                  lw_pk, HCD, CD);

    // xl = h @ Wl + bl ; xr = h @ Wr + br   (WMMA bf16)
    wmma_gemm_bias_kernel<<<gemmBlocks(NODES, HCD), T, 0, stream>>>(h_bf, wl_pk, bl + i * HCD, xl,
                                                                    NODES, CD, HCD);
    wmma_gemm_bias_kernel<<<gemmBlocks(NODES, HCD), T, 0, stream>>>(h_bf, wr_pk, br + i * HCD, xr,
                                                                    NODES, CD, HCD);

    // reset segment buffers + aggregation target + reduction accumulators
    fill_u32_kernel<<<blocksFor((long)NODES * HEADS, T), T, 0, stream>>>(mbuf, ENC_NEG_INF,
                                                                         (long)NODES * HEADS);
    fill_f32_kernel<<<blocksFor((long)NODES * HEADS, T), T, 0, stream>>>(sbuf, 0.0f,
                                                                         (long)NODES * HEADS);
    fill_f32_kernel<<<blocksFor((long)NODES * HCD, T), T, 0, stream>>>(agg, 0.0f,
                                                                       (long)NODES * HCD);
    fill_f32_kernel<<<1, 32, 0, stream>>>((float*)red, 0.0f, 4);

    // edge pipeline: logits+max, exp+sum, scatter (all L2-resident working set)
    edge_logits_kernel<<<blocksFor(edgeThreads, T), T, 0, stream>>>(
        xl, xr, eattr, srcIdx, dstIdx, We + (size_t)i * HCD, att + (size_t)i * HEADS * CD,
        logits, mbuf);
    edge_exp_kernel<<<blocksFor(edgeThreads, T), T, 0, stream>>>(logits, mbuf, dstIdx, sbuf);
    edge_scatter_kernel<<<blocksFor(edgeThreads, T), T, 0, stream>>>(logits, sbuf, xl, srcIdx,
                                                                     dstIdx, agg);

    // graph LayerNorm over all N*HC values (+gat_b), ReLU, emit bf16 for lin GEMM
    reduce_sumsq_kernel<<<2048, T, 0, stream>>>(agg, gat_b + i * HCD, red, (long)NODES * HCD);
    graph_ln_relu_kernel<<<blocksFor((long)NODES * HCD, T), T, 0, stream>>>(
        agg, gat_b + i * HCD, red, ln1_w + i * HCD, ln1_b + i * HCD, g_bf, (long)NODES * HCD);

    // g @ lin_w + lin_b (WMMA), node LN, ReLU, JK max into out
    wmma_gemm_bias_kernel<<<gemmBlocks(NODES, CD), T, 0, stream>>>(g_bf, lw_pk, lin_b + i * CD,
                                                                   tmpC, NODES, HCD, CD);
    node_ln_relu_max_kernel<<<(NODES + 7) / 8, T, 0, stream>>>(tmpC, ln2_w + i * CD,
                                                               ln2_b + i * CD, h, h_bf, out,
                                                               NODES, 0);
  }
}
